// QuantLinearINT4_35338990912096
// MI455X (gfx1250) — compile-verified, matrix operand order checked by reading, not run on hardware
//
#include <hip/hip_runtime.h>
#include <stdint.h>

// ---------------------------------------------------------------------------
// QuantLinear INT4 (GPTQ-style) for gfx1250 (MI455X)
//
//   out[m,n] = sum_k x[m,k] * (q[n,k]*sc[n] + zr[n]) + bias[n]
//            = sc[n]*ACC[m,n] + (zr[n] - 128*sc[n])*rowsum_x[m] + bias[n]
//   where ACC = X_bf16 * QB^T,  QB[n,k] = 128 + q[n,k]  (bf16 = 0x4300 | q,
//   built with pure bit ops -- no int->float->bf16 arithmetic in the loop).
//
//   Pass 1: X f32 -> bf16 once into workspace
//   Pass 2: rowsum_x[m] = sum_k X_bf16[m,k]
//   Pass 3: bf16 WMMA GEMM (256x128 block tile, 16 waves, 108KB LDS —
//           leveraging CDNA5's 320KB WGP LDS); A staged via async
//           global->LDS DMA (ASYNCcnt), W nibbles spliced to bf16 with
//           v_perm/v_and_or, epilogue fixup.
// ---------------------------------------------------------------------------

#define M_DIM 4096
#define K_DIM 4096
#define N_DIM 11008
#define PACK  8

typedef __attribute__((ext_vector_type(16))) __bf16 v16bf;
typedef __attribute__((ext_vector_type(8)))  __bf16 v8bf;
typedef __attribute__((ext_vector_type(8)))  float  v8f;
typedef int vint4 __attribute__((vector_size(16)));   // matches builtin param

#define AS1 __attribute__((address_space(1)))
#define AS3 __attribute__((address_space(3)))

constexpr int THREADS = 512;        // 16 waves: 4 (M) x 4 (N)
constexpr int BM  = 256;            // block tile M (amortizes dequant VALU)
constexpr int BN  = 128;            // block tile N
constexpr int BK  = 64;             // two WMMA k-steps per stage
constexpr int LDA = BK + 8;         // padded LDS row stride (144B, 16B aligned)
constexpr int KT  = K_DIM / BK;     // 64 iterations

#if __has_builtin(__builtin_amdgcn_global_load_async_to_lds_b128)
#define USE_ASYNC_LDS 1
#else
#define USE_ASYNC_LDS 0
#warning "gfx1250 async-to-LDS builtin unavailable; using synchronous staging"
#endif

__device__ __forceinline__ void wait_async0() {
#if USE_ASYNC_LDS
#if __has_builtin(__builtin_amdgcn_s_wait_asynccnt)
  __builtin_amdgcn_s_wait_asynccnt(0);
#else
  asm volatile("s_wait_asynccnt 0" ::: "memory");
#endif
#endif
}

// Two bf16 lanes {0x4300|e_b, 0x4300|o_b} from byte b of e/o nibble planes.
__device__ __forceinline__ unsigned qpair(unsigned e, unsigned o, int b) {
#if __has_builtin(__builtin_amdgcn_perm)
  unsigned sel = (unsigned)(b | (b << 8) | ((4 + b) << 16) | ((4 + b) << 24));
  unsigned t = __builtin_amdgcn_perm(o, e, sel);        // {e_b, e_b, o_b, o_b}
  return (t & 0x00FF00FFu) | 0x43004300u;               // v_and_or_b32
#else
  unsigned eb = (e >> (8 * b)) & 0xFFu;
  unsigned ob = (o >> (8 * b)) & 0xFFu;
  return eb | (ob << 16) | 0x43004300u;
#endif
}

// ---------------------------------------------------------------------------
// Pass 1: convert X (f32) to bf16 in workspace, 8 elements/thread.
// ---------------------------------------------------------------------------
__global__ __launch_bounds__(256) void
convert_x_bf16_kernel(const float* __restrict__ x, unsigned short* __restrict__ xb) {
  size_t idx = (size_t)blockIdx.x * 256 + threadIdx.x;   // 0 .. M*K/8-1
  const float4* p = (const float4*)x + idx * 2;
  float4 a = p[0];
  float4 b = p[1];
  v8bf o;
  o[0] = (__bf16)a.x; o[1] = (__bf16)a.y; o[2] = (__bf16)a.z; o[3] = (__bf16)a.w;
  o[4] = (__bf16)b.x; o[5] = (__bf16)b.y; o[6] = (__bf16)b.z; o[7] = (__bf16)b.w;
  *((v8bf*)xb + idx) = o;
}

// ---------------------------------------------------------------------------
// Pass 2: per-row sums of X_bf16 (f32 accumulate); one block per row.
// ---------------------------------------------------------------------------
__global__ __launch_bounds__(256) void
rowsum_kernel(const unsigned short* __restrict__ xb, float* __restrict__ rsx) {
  __shared__ float red[256];
  const int row = blockIdx.x;
  const int t   = threadIdx.x;
  const v8bf* p = (const v8bf*)(const void*)(xb + (size_t)row * K_DIM) + t * 2;
  v8bf a = p[0];
  v8bf b = p[1];
  float s = 0.0f;
  #pragma unroll
  for (int i = 0; i < 8; ++i) s += (float)a[i] + (float)b[i];
  red[t] = s;
  __syncthreads();
  #pragma unroll
  for (int off = 128; off > 0; off >>= 1) {
    if (t < off) red[t] += red[t + off];
    __syncthreads();
  }
  if (t == 0) rsx[row] = red[0];
}

// ---------------------------------------------------------------------------
// Pass 3: GEMM. 512 threads = 16 waves (4 M x 4 N); wave computes 64x32.
// ---------------------------------------------------------------------------
__global__ __launch_bounds__(THREADS) void
quantlinear_int4_wmma_kernel(const unsigned short* __restrict__ xb,  // bf16 X
                             const int*   __restrict__ wq,
                             const float* __restrict__ scales,
                             const float* __restrict__ zeros,
                             const float* __restrict__ bias,
                             const float* __restrict__ rsx,
                             float*       __restrict__ out) {
  __shared__ __align__(16) unsigned short sA[2][BM * LDA];  // bf16 X tiles
  __shared__ __align__(16) unsigned short sB[2][BN * LDA];  // bf16 (128+q) tiles

  const int tid  = threadIdx.x;
  const int lane = tid & 31;
  const int wv   = tid >> 5;     // 0..15
  const int wm   = wv & 3;       // wave M slot: 4 x 64 rows
  const int wn   = wv >> 2;      // wave N slot: 4 x 32 cols
  const int lm   = lane & 15;
  const int lh   = lane >> 4;

  const int n0 = blockIdx.x * BN;
  const int m0 = blockIdx.y * BM;

  // AS3 base for async DMA destinations
  AS3 unsigned short* sA3 =
      (AS3 unsigned short*)(uint32_t)(uintptr_t)&sA[0][0];

  // ---- staging roles -----------------------------------------------------
  // A: 256 rows x 8 chunks(16B) = 2048 chunks; 4 per thread
  // B: 128 rows x 8 int32 = 1024 int32; thread: row nl = t>>2, int32s p0..p0+1
  const int nl = tid >> 2;
  const int p0 = (tid & 3) * 2;
  const int* wrow = wq + (size_t)(n0 + nl) * (K_DIM / PACK);

  v8f acc[8];                        // [im*2+in], im=0..3, in=0..1
  #pragma unroll
  for (int i = 0; i < 8; ++i) acc[i] = (v8f)0.0f;

#if !USE_ASYNC_LDS
  v8bf aregs[4];
#endif
  int2 breg;

  // ---- A staging: async DMA straight into LDS (no VGPR data path) --------
  auto stage_a = [&](int buf, int kt) {
    #pragma unroll
    for (int p = 0; p < 4; ++p) {
      int idx = p * THREADS + tid;
      int row = idx >> 3;
      int c8  = idx & 7;
      const unsigned short* g =
          xb + (size_t)(m0 + row) * K_DIM + kt * BK + c8 * 8;
#if USE_ASYNC_LDS
      AS3 unsigned short* l = sA3 + buf * (BM * LDA) + row * LDA + c8 * 8;
      __builtin_amdgcn_global_load_async_to_lds_b128(
          (AS1 vint4*)(uintptr_t)g, (AS3 vint4*)l, 0, 0);
#else
      aregs[p] = *(const v8bf*)(const void*)g;
#endif
    }
  };

#if !USE_ASYNC_LDS
  auto store_a = [&](int buf) {
    #pragma unroll
    for (int p = 0; p < 4; ++p) {
      int idx = p * THREADS + tid;
      int row = idx >> 3;
      int c8  = idx & 7;
      *(v8bf*)(void*)&sA[buf][row * LDA + c8 * 8] = aregs[p];
    }
  };
#endif

  auto fetch_b = [&](int kt) {
    breg = *(const int2*)(wrow + kt * (BK / PACK) + p0);
  };

  // int4 nibbles -> bf16(128+q) via pure bit splicing (no cvt/fma)
  auto stage_b = [&](int buf) {
    #pragma unroll
    for (int q = 0; q < 2; ++q) {
      unsigned w = (q == 0) ? (unsigned)breg.x : (unsigned)breg.y;
      unsigned e = w & 0x0F0F0F0Fu;          // nibbles 0,2,4,6
      unsigned o = (w >> 4) & 0x0F0F0F0Fu;   // nibbles 1,3,5,7
      uint4 d;
      d.x = qpair(e, o, 0);
      d.y = qpair(e, o, 1);
      d.z = qpair(e, o, 2);
      d.w = qpair(e, o, 3);
      *(uint4*)(void*)&sB[buf][nl * LDA + (p0 + q) * 8] = d;
    }
  };

  // ---- prologue ----------------------------------------------------------
  stage_a(0, 0);
  fetch_b(0);
  stage_b(0);
#if !USE_ASYNC_LDS
  store_a(0);
#endif
  wait_async0();
  __syncthreads();

  // ---- main loop ---------------------------------------------------------
  for (int kt = 0; kt < KT; ++kt) {
    const int cur = kt & 1;
    const bool has_next = (kt + 1) < KT;

    if (has_next) {
      stage_a(cur ^ 1, kt + 1);
      fetch_b(kt + 1);
      if (kt + 2 < KT) {
        __builtin_prefetch(wrow + (kt + 2) * (BK / PACK), 0, 1);
      }
    }

    const unsigned short* a_ = sA[cur];
    const unsigned short* b_ = sB[cur];

    #pragma unroll
    for (int ks = 0; ks < 2; ++ks) {           // two WMMA k-steps per stage
      v16bf afrag[4];
      #pragma unroll
      for (int im = 0; im < 4; ++im) {
        const unsigned short* p =
            a_ + (wm * 64 + im * 16 + lm) * LDA + ks * 32 + lh * 8;
        v8bf flo = *(const v8bf*)(const void*)p;          // ds_load_b128
        v8bf fhi = *(const v8bf*)(const void*)(p + 16);   // ds_load_b128
        afrag[im] = __builtin_shufflevector(flo, fhi,
            0,1,2,3,4,5,6,7,8,9,10,11,12,13,14,15);
      }
      v16bf bfrag[2];
      #pragma unroll
      for (int in = 0; in < 2; ++in) {
        const unsigned short* p =
            b_ + (wn * 32 + in * 16 + lm) * LDA + ks * 32 + lh * 8;
        v8bf flo = *(const v8bf*)(const void*)p;
        v8bf fhi = *(const v8bf*)(const void*)(p + 16);
        bfrag[in] = __builtin_shufflevector(flo, fhi,
            0,1,2,3,4,5,6,7,8,9,10,11,12,13,14,15);
      }
      #pragma unroll
      for (int im = 0; im < 4; ++im) {
        #pragma unroll
        for (int in = 0; in < 2; ++in) {
          acc[im * 2 + in] = __builtin_amdgcn_wmma_f32_16x16x32_bf16(
              false, afrag[im], false, bfrag[in],
              (short)0, acc[im * 2 + in], false, false);
        }
      }
    }

    if (has_next) {
      stage_b(cur ^ 1);
#if !USE_ASYNC_LDS
      store_a(cur ^ 1);
#endif
    }
    wait_async0();
    __syncthreads();
  }

  // ---- epilogue: out = sc*acc + (zr-128*sc)*rowsum_x + bias --------------
  #pragma unroll
  for (int im = 0; im < 4; ++im) {
    const int mbase = m0 + wm * 64 + im * 16 + lh * 8;
    float rs[8];
    #pragma unroll
    for (int r = 0; r < 8; ++r) rs[r] = rsx[mbase + r];
    #pragma unroll
    for (int in = 0; in < 2; ++in) {
      const int n = n0 + wn * 32 + in * 16 + lm;
      const float sc = scales[n];
      const float czr = zeros[n] - 128.0f * sc;
      const float bv = bias[n];
      v8f c = acc[im * 2 + in];
      #pragma unroll
      for (int r = 0; r < 8; ++r) {
        out[(size_t)(mbase + r) * N_DIM + n] = fmaf(sc, c[r], fmaf(czr, rs[r], bv));
      }
    }
  }
}

extern "C" void kernel_launch(void* const* d_in, const int* in_sizes, int n_in,
                              void* d_out, int out_size, void* d_ws, size_t ws_size,
                              hipStream_t stream) {
  const float* x      = (const float*)d_in[0];
  const int*   wq     = (const int*)d_in[1];
  const float* scales = (const float*)d_in[2];
  const float* zeros  = (const float*)d_in[3];
  const float* bias   = (const float*)d_in[4];
  float*       out    = (float*)d_out;

  unsigned short* xb  = (unsigned short*)d_ws;     // M*K bf16 = 32 MiB
  float* rsx = (float*)((char*)d_ws + (size_t)M_DIM * K_DIM * 2);  // +16 KiB

  // Pass 1: one-time f32 -> bf16 conversion of activations
  {
    int total = (M_DIM * K_DIM) / 8;
    convert_x_bf16_kernel<<<dim3(total / 256), dim3(256), 0, stream>>>(x, xb);
  }
  // Pass 2: row sums of X_bf16 for the affine-zero correction
  rowsum_kernel<<<dim3(M_DIM), dim3(256), 0, stream>>>(xb, rsx);

  // Pass 3: WMMA GEMM with bit-spliced int4 -> bf16(128+q) weights
  {
    dim3 grid(N_DIM / BN, M_DIM / BM);             // (86, 16)
    quantlinear_int4_wmma_kernel<<<grid, dim3(THREADS), 0, stream>>>(
        xb, wq, scales, zeros, bias, rsx, out);
  }
}
